// BetweennessModule_34333968564859
// MI455X (gfx1250) — matrix operand order repaired and checked by reading, not compiled, not run on hardware
//
#include <hip/hip_runtime.h>

typedef __attribute__((ext_vector_type(16))) __bf16 v16bf;
typedef __attribute__((ext_vector_type(8)))  __bf16 v8bf;
typedef __attribute__((ext_vector_type(8)))  float  v8f;
typedef __attribute__((ext_vector_type(4)))  float  v4f;

#define S_LEN     4096
#define D_DIM     1024
#define ROWS_OUT  30          // score positions produced per workgroup
#define M_TILE    32          // content rows per workgroup (ROWS_OUT + 2 halo)
#define LDA_STRIDE 1032       // halves: 1024 + 8 pad -> 2064B rows (16B aligned, bank-tiling)
#define LDC_STRIDE 260        // floats: 256 + 4 pad
#define ADJ_SCALE 0.1f
#define EPS_V     1e-6f

// ---- Prepass: W [e][d] f32  ->  Wp[d/32][e][d%32] bf16 (B-fragment friendly) ----
__global__ __launch_bounds__(256) void wconv_kernel(const float* __restrict__ W,
                                                    unsigned short* __restrict__ Wp) {
  int o  = blockIdx.x * 256 + threadIdx.x;    // 1M outputs
  int j  = o & 31;                            // d % 32
  int e  = (o >> 5) & 1023;                   // column (N) index
  int kc = o >> 15;                           // d / 32
  union { __bf16 h; unsigned short u; } cv;
  cv.h = (__bf16)W[e * D_DIM + kc * 32 + j];
  Wp[o] = cv.u;
}

// ---- Fused GEMM (bf16 WMMA) + neighbor-distance betweenness ----
__global__ __launch_bounds__(256) void betweenness_kernel(const float* __restrict__ x,
                                                          const unsigned short* __restrict__ Wp,
                                                          const float* __restrict__ gate,
                                                          float* __restrict__ out) {
  __shared__ unsigned short lds_a[M_TILE * LDA_STRIDE];   // full bf16 A tile (32 x 1024)
  __shared__ float lds_c[M_TILE * LDC_STRIDE];            // f32 content slab (32 x 256)
  __shared__ float sm_d1[32];                             // sum sq diffs (i, i+1)
  __shared__ float sm_d2[32];                             // sum sq diffs (i, i+2)

  const int tid  = threadIdx.x;
  const int lane = tid & 31;
  const int w    = tid >> 5;                 // wave id 0..7
  const int bz   = blockIdx.y;               // batch
  const int r0   = blockIdx.x * ROWS_OUT;    // first content row of this tile

  if (tid < 32) { sm_d1[tid] = 0.0f; sm_d2[tid] = 0.0f; }

  const float* xb = x + (size_t)bz * S_LEN * D_DIM;

  // ---- stage full A tile ONCE: x f32 -> bf16 -> LDS (32 rows x 1024 K) ----
  {
    int row = tid >> 3;
    int c0  = (tid & 7) << 3;                 // 8 contiguous elements per thread per step
    int g   = r0 + row; if (g > S_LEN - 1) g = S_LEN - 1;
    const float* src = xb + (size_t)g * D_DIM;
    unsigned short* dst = lds_a + row * LDA_STRIDE;
#pragma unroll
    for (int cc = 0; cc < D_DIM; cc += 64) {
      v4f s0 = *(const v4f*)(src + cc + c0);
      v4f s1 = *(const v4f*)(src + cc + c0 + 4);
      union { v8bf h; struct { unsigned long long a, b; } u; } pk;
      pk.h[0] = (__bf16)s0.x; pk.h[1] = (__bf16)s0.y;
      pk.h[2] = (__bf16)s0.z; pk.h[3] = (__bf16)s0.w;
      pk.h[4] = (__bf16)s1.x; pk.h[5] = (__bf16)s1.y;
      pk.h[6] = (__bf16)s1.z; pk.h[7] = (__bf16)s1.w;
      *(v8bf*)(dst + cc + c0) = pk.h;
    }
  }
  __syncthreads();

  const int mrow = lane & 15;                     // M (and N) index within tile
  const int aoff = (lane < 16) ? 0 : 16;          // A: half-lane K offset (halves)
  const int boff = (lane < 16) ? 0 : 16;          // B: half-lane K offset (halves)

  const unsigned short* ap0 = lds_a + mrow * LDA_STRIDE + aoff;        // rows 0..15
  const unsigned short* ap1 = lds_a + (mrow + 16) * LDA_STRIDE + aoff; // rows 16..31

  for (int nit = 0; nit < 4; ++nit) {             // 4 x 256 = 1024 columns
    v8f acc00 = {};                               // rows 0..15,  cols +0
    v8f acc01 = {};                               // rows 0..15,  cols +16
    v8f acc10 = {};                               // rows 16..31, cols +0
    v8f acc11 = {};                               // rows 16..31, cols +16
    const int ncol0 = nit * 256 + w * 32 + mrow;  // lane's global columns e
    const int ncol1 = ncol0 + 16;

    // ---- barrier-free K loop: 32 steps, 4 WMMAs each ----
#pragma unroll 2
    for (int kk = 0; kk < D_DIM; kk += 32) {
      const size_t kblk = (size_t)(kk >> 5) * 1024;
      v16bf b0 = *(const v16bf*)(Wp + (kblk + ncol0) * 32 + boff);
      v16bf b1 = *(const v16bf*)(Wp + (kblk + ncol1) * 32 + boff);

      union { v16bf v; v8bf h[2]; } a0, a1;
      a0.h[0] = *(const v8bf*)(ap0 + kk);
      a0.h[1] = *(const v8bf*)(ap0 + kk + 8);
      a1.h[0] = *(const v8bf*)(ap1 + kk);
      a1.h[1] = *(const v8bf*)(ap1 + kk + 8);

      acc00 = __builtin_amdgcn_wmma_f32_16x16x32_bf16(false, a0.v, false, b0,
                                                      (short)0, acc00, false, false);
      acc01 = __builtin_amdgcn_wmma_f32_16x16x32_bf16(false, a0.v, false, b1,
                                                      (short)0, acc01, false, false);
      acc10 = __builtin_amdgcn_wmma_f32_16x16x32_bf16(false, a1.v, false, b0,
                                                      (short)0, acc10, false, false);
      acc11 = __builtin_amdgcn_wmma_f32_16x16x32_bf16(false, a1.v, false, b1,
                                                      (short)0, acc11, false, false);
    }

    __syncthreads();   // previous slab's diff readers are done with lds_c

    // ---- dump accumulators (C layout: lane = column, VGPR v = row rb+v) ----
    {
      int nl0 = w * 32 + mrow;
      int nl1 = nl0 + 16;
      int rb  = (lane < 16) ? 0 : 8;
#pragma unroll
      for (int v = 0; v < 8; ++v) {
        lds_c[(rb + v) * LDC_STRIDE + nl0]        = acc00[v];
        lds_c[(rb + v) * LDC_STRIDE + nl1]        = acc01[v];
        lds_c[(16 + rb + v) * LDC_STRIDE + nl0]   = acc10[v];
        lds_c[(16 + rb + v) * LDC_STRIDE + nl1]   = acc11[v];
      }
    }
    __syncthreads();

    // ---- accumulate squared neighbor distances over this 256-column slab ----
    {
      int r  = tid & 31;
      int c0 = (tid >> 5) * 32;
      if (r < 31) {
        float p = 0.0f;
#pragma unroll
        for (int c = 0; c < 32; ++c) {
          float d = lds_c[(r + 1) * LDC_STRIDE + c0 + c] - lds_c[r * LDC_STRIDE + c0 + c];
          p += d * d;
        }
        atomicAdd(&sm_d1[r], p);        // ds_add_f32
      }
      if (r < 30) {
        float q = 0.0f;
#pragma unroll
        for (int c = 0; c < 32; ++c) {
          float d = lds_c[(r + 2) * LDC_STRIDE + c0 + c] - lds_c[r * LDC_STRIDE + c0 + c];
          q += d * d;
        }
        atomicAdd(&sm_d2[r], q);
      }
    }
    // next nit's first __syncthreads() orders lds_c reuse
  }
  __syncthreads();

  // ---- finalize: sqrt, path/direct, relu score, gate ----
  const float gate_val = gate[0] * 0.5f;
  if (tid < ROWS_OUT) {
    int iglob = r0 + tid;                       // score index in [0, S-2)
    if (iglob < S_LEN - 2) {
      float d1a  = sqrtf(sm_d1[tid]);
      float d1b  = sqrtf(sm_d1[tid + 1]);
      float dir  = sqrtf(sm_d2[tid]);
      float path = d1a + d1b;
      float score = 1.0f - (path - dir) / fmaxf(dir, EPS_V);
      score = fmaxf(score, 0.0f);
      float bet = score * (1.0f / (float)(S_LEN - 2));
      out[(size_t)bz * S_LEN + iglob + 1] = gate_val * (bet - 0.5f) * ADJ_SCALE;
    }
  }
  if (blockIdx.x == 0 && tid < 2) {             // betweenness == 0 at the boundaries
    int p = (tid == 0) ? 0 : (S_LEN - 1);
    out[(size_t)bz * S_LEN + p] = gate_val * (-0.5f) * ADJ_SCALE;
  }
}

extern "C" void kernel_launch(void* const* d_in, const int* in_sizes, int n_in,
                              void* d_out, int out_size, void* d_ws, size_t ws_size,
                              hipStream_t stream) {
  const float* x    = (const float*)d_in[0];          // [8, 4096, 1024]
  const float* W    = (const float*)d_in[1];          // [1024, 1024]
  // d_in[2] = bias: cancels exactly in neighbor differences -> unused
  const float* gate = (const float*)d_in[3];          // [1]
  float* out = (float*)d_out;                         // [8, 4096]

  unsigned short* Wp = (unsigned short*)d_ws;         // 2 MB bf16 W, K-blocked

  wconv_kernel<<<(D_DIM * D_DIM) / 256, 256, 0, stream>>>(W, Wp);

  dim3 grid((S_LEN - 2 + ROWS_OUT - 1) / ROWS_OUT, 8);
  betweenness_kernel<<<grid, 256, 0, stream>>>(x, Wp, gate, out);

  (void)in_sizes; (void)n_in; (void)out_size; (void)ws_size;
}
